// MemorizingTransformerLayer_89945205113177
// MI455X (gfx1250) — matrix operand order repaired and checked by reading, not compiled, not run on hardware
//
#include <hip/hip_runtime.h>

// ---------------------------------------------------------------------------
// Problem constants (from reference)
// ---------------------------------------------------------------------------
#define B_ 4
#define S_ 256
#define E_ 1024
#define H_ 16
#define HD_ 64
#define MM_ 65536
#define KK_ 32
#define FF_ 4096
#define NTOK (B_ * S_)     // 1024 tokens
#define SK_ (S_ * KK_)     // 8192 retrieved rows per batch

// GEMM tiling. Requirements (hold at every call site):
//   Md % TM == 0, Kd % TK == 0, Kd/TK >= 2
//   BT==true  -> Nd % TN == 0 (no N guards at all)
//   BT==false -> Nd % 4 == 0  (float4-granular N guard)
#define TM 64
#define TN 128
#define TK 32
#define NPAIR (TK / 2)      // 16 k-pairs per tile
#define LDA_S (TK + 2)      // 34 halfs per A row (even -> 4B-aligned pairs)
#define LDB2 (TN + 4)       // 132 u32 per B pair-row (x4B = 16B-aligned rows)

typedef __attribute__((ext_vector_type(8))) float v8f;
typedef __attribute__((ext_vector_type(2))) float f32x2;

// ---------------------------------------------------------------------------
// 16-bit matrix element type for WMMA: prefer bf16 (range-safe for fp32 data),
// fall back to the codegen-verified f16 builtin if bf16 is unavailable.
// Conversions go through __builtin_convertvector -> v_cvt_pk_bf16_f32.
// ---------------------------------------------------------------------------
#if __has_builtin(__builtin_amdgcn_wmma_f32_16x16x32_bf16)
typedef __attribute__((ext_vector_type(16))) __bf16 frag16_t;
typedef __attribute__((ext_vector_type(2)))  __bf16 e16x2;
#define WMMA_MAC(Af, Bf, Cf) \
  __builtin_amdgcn_wmma_f32_16x16x32_bf16(false, (Af), false, (Bf), (short)0, (Cf), false, false)
#else
typedef __attribute__((ext_vector_type(16))) _Float16 frag16_t;
typedef __attribute__((ext_vector_type(2)))  _Float16 e16x2;
#define WMMA_MAC(Af, Bf, Cf) \
  __builtin_amdgcn_wmma_f32_16x16x32_f16(false, (Af), false, (Bf), (short)0, (Cf), false, false)
#endif

union Frag16 { frag16_t v; unsigned int u[8]; };

// pack two f32 into one u32 of 16-bit elements (lo = first), hardware cvt
__device__ inline unsigned int pack2v(float a, float b) {
  f32x2 v; v.x = a; v.y = b;
  e16x2 r = __builtin_convertvector(v, e16x2);
  unsigned int u; __builtin_memcpy(&u, &r, 4);
  return u;
}

// ---------------------------------------------------------------------------
// Tiled WMMA GEMM:  C[z] = act( scale * A[z] * B[z] + bias )
//   A: Md x Kd (row-major f32), lda
//   B: BT==false -> Kd x Nd row-major f32; BT==true -> Nd x Kd row-major
//      (B^T math), element type TB in {float, unsigned short(=bf16 pairs)}
//   z-batching: z = zb*Hz + zh;  base += zb*s?b + zh*s?h
//   ACT: 0 = none, 1 = exact GELU
//
// Block: 256 threads = 8 waves (wave32); 2x4 wave grid, each wave a 32x32
// patch (2x2 WMMA 16x16 tiles); block tile 64x128; K-step 32.
// Software pipeline with REGISTER staging + double-buffered LDS:
//   storeTile(cur)  : s_wait on last iteration's global loads, cvt_pk, DS store
//   loadTile(next)  : issue next tile's global loads (no wait)
//   barrier; compute(cur): 16 ds_load_b32-class reads + 4 WMMAs
// so global-load latency is hidden behind a barrier + WMMA block, not waited
// on element-by-element. Last iteration peeled -> branch-free steady state.
// LDS B layout is pre-paired: Bs[pair p][n] = u32{16b K=2p, 16b K=2p+1},
// so each B fragment VGPR is one ds_load_b32.
// ---------------------------------------------------------------------------
template <bool BT, typename TB, int ACT>
__global__ __launch_bounds__(256) void gemm_wmma(
    const float* __restrict__ A, int lda, long long sAb, long long sAh,
    const TB* __restrict__ Bm, int ldb, long long sBb, long long sBh,
    float* __restrict__ C, int ldc, long long sCb, long long sCh,
    int Md, int Nd, int Kd, int Hz,
    const float* __restrict__ bias, float scale) {
  const int z = blockIdx.z;
  const int zb = z / Hz, zh = z - zb * Hz;
  A  += (long long)zb * sAb + (long long)zh * sAh;
  Bm += (long long)zb * sBb + (long long)zh * sBh;
  C  += (long long)zb * sCb + (long long)zh * sCh;

  __shared__ __attribute__((aligned(16))) unsigned short As[2 * TM * LDA_S];
  __shared__ __attribute__((aligned(16))) unsigned int   Bs[2 * NPAIR * LDB2];

  const int m0 = blockIdx.y * TM;
  const int n0 = blockIdx.x * TN;
  const int t = threadIdx.x;
  const int lane = t & 31;
  const int wave = t >> 5;
  const int wm = wave >> 2;   // 0..1
  const int wn = wave & 3;    // 0..3
  const int half = lane >> 4;
  const int l15 = lane & 15;

  constexpr bool EPACK = (sizeof(TB) == 2);  // BT operand already 16-bit pairs

  v8f acc[2][2];
  for (int i = 0; i < 2; ++i)
    for (int j = 0; j < 2; ++j)
      for (int e = 0; e < 8; ++e) acc[i][j][e] = 0.0f;

  // ---- register staging buffers (live across pipeline stages) ----
  f32x2 aReg[4];
  f32x2 btf[8];            // BT, f32 source
  unsigned int btu[8];     // BT, 16-bit source
  float4 bn0[2], bn1[2];   // non-BT: rows k and k+1

  // issue the global loads for one K-tile (no waits implied here)
  auto loadTile = [&](int k0) {
#pragma unroll
    for (int i = 0; i < 4; ++i) {          // A: 64 rows x 16 pairs / 256 thr
      int e = i * 256 + t;
      int r = e >> 4, p = e & 15;
      aReg[i] = *(const f32x2*)(A + (long long)(m0 + r) * lda + k0 + 2 * p);
    }
    if (BT) {
#pragma unroll
      for (int i = 0; i < 8; ++i) {        // 16 pairs x 128 n / 256 thr
        int e = i * 256 + t;
        int p = e & 15, nn = e >> 4;
        int gn = n0 + nn;                  // BT => Nd % TN == 0: no guard
        if (EPACK)
          btu[i] = *(const unsigned int*)((const unsigned short*)Bm +
                                          (long long)gn * ldb + k0 + 2 * p);
        else
          btf[i] = *(const f32x2*)((const float*)Bm +
                                   (long long)gn * ldb + k0 + 2 * p);
      }
    } else {
#pragma unroll
      for (int i = 0; i < 2; ++i) {        // 16 pairs x 32 n-quads / 256 thr
        int e = i * 256 + t;
        int p = e >> 5, q = e & 31;
        int nn = q * 4;
        int gn = n0 + nn;
        float4 zz; zz.x = zz.y = zz.z = zz.w = 0.0f;
        if (gn < Nd) {
          bn0[i] = *(const float4*)((const float*)Bm + (long long)(k0 + 2 * p) * ldb + gn);
          bn1[i] = *(const float4*)((const float*)Bm + (long long)(k0 + 2 * p + 1) * ldb + gn);
        } else { bn0[i] = zz; bn1[i] = zz; }
      }
    }
  };

  // convert + store the register-staged tile into LDS buffer `buf`
  auto storeTile = [&](int buf) {
    unsigned short* Ad = As + buf * (TM * LDA_S);
    unsigned int*   Bd = Bs + buf * (NPAIR * LDB2);
#pragma unroll
    for (int i = 0; i < 4; ++i) {
      int e = i * 256 + t;
      int r = e >> 4, p = e & 15;
      *(unsigned int*)&Ad[r * LDA_S + 2 * p] = pack2v(aReg[i].x, aReg[i].y);
    }
    if (BT) {
#pragma unroll
      for (int i = 0; i < 8; ++i) {
        int e = i * 256 + t;
        int p = e & 15, nn = e >> 4;
        Bd[p * LDB2 + nn] = EPACK ? btu[i] : pack2v(btf[i].x, btf[i].y);
      }
    } else {
#pragma unroll
      for (int i = 0; i < 2; ++i) {
        int e = i * 256 + t;
        int p = e >> 5, q = e & 31;
        int nn = q * 4;
        uint4 w4;
        w4.x = pack2v(bn0[i].x, bn1[i].x);
        w4.y = pack2v(bn0[i].y, bn1[i].y);
        w4.z = pack2v(bn0[i].z, bn1[i].z);
        w4.w = pack2v(bn0[i].w, bn1[i].w);
        *(uint4*)&Bd[p * LDB2 + nn] = w4;
      }
    }
  };

  // ---- consume one LDS buffer: 16 ds_load_b32-class reads, 4 WMMAs ----
  auto compute = [&](int buf) {
    const unsigned short* Asb = As + buf * (TM * LDA_S);
    const unsigned int*   Bsb = Bs + buf * (NPAIR * LDB2);
    Frag16 afr[2], bfr[2];
#pragma unroll
    for (int tm = 0; tm < 2; ++tm) {
      int m = wm * 32 + tm * 16 + l15;    // A: M = lane&15 (both halves)
#pragma unroll
      for (int vg = 0; vg < 8; ++vg) {
        // A 16x32 layout: VGPR vg holds K-pair (vg&3)*2 + (vg>>2)*16 + half*8
        int k = ((vg & 3) << 1) + ((vg >> 2) << 4) + (half << 3);
        afr[tm].u[vg] = *(const unsigned int*)&Asb[m * LDA_S + k];
      }
    }
#pragma unroll
    for (int tn = 0; tn < 2; ++tn) {
      int n = wn * 32 + tn * 16 + l15;    // B: N = lane&15
#pragma unroll
      for (int vg = 0; vg < 8; ++vg) {
        // B 32x16 layout: VGPR vg holds K-pair p = (vg&3) + (vg>>2)*8 + half*4 (one b32 read)
        bfr[tn].u[vg] = Bsb[((vg & 3) + ((vg >> 2) << 3) + (half << 2)) * LDB2 + n];
      }
    }
#pragma unroll
    for (int tm = 0; tm < 2; ++tm)
#pragma unroll
      for (int tn = 0; tn < 2; ++tn)
        acc[tm][tn] = WMMA_MAC(afr[tm].v, bfr[tn].v, acc[tm][tn]);
  };

  const int nk = Kd / TK;                 // >= 2 at every call site
  loadTile(0);
  int cur = 0;
  for (int ki = 0; ki < nk - 1; ++ki) {   // branch-free steady state
    storeTile(cur);                       // waits prior loads; cvt_pk; DS store
    loadTile((ki + 1) * TK);              // next tile's loads go in flight
    __syncthreads();
    compute(cur);                         // WMMAs overlap the in-flight loads
    cur ^= 1;
  }
  storeTile(cur);                         // peeled last tile
  __syncthreads();
  compute(cur);

  // ---- epilogue: D layout -> VGPR i: lanes0-15 M=i, lanes16-31 M=i+8 ----
#pragma unroll
  for (int tm = 0; tm < 2; ++tm) {
#pragma unroll
    for (int tn = 0; tn < 2; ++tn) {
      int ncol = n0 + wn * 32 + tn * 16 + l15;
      if (ncol < Nd) {
        float badd = bias ? bias[ncol] : 0.0f;
        int mbase = m0 + wm * 32 + tm * 16 + half * 8;
#pragma unroll
        for (int i2 = 0; i2 < 8; ++i2) {
          float vv = acc[tm][tn][i2] * scale + badd;
          if (ACT == 1) vv = 0.5f * vv * (1.0f + erff(vv * 0.70710678118654752f));
          C[(long long)(mbase + i2) * ldc + ncol] = vv;
        }
      }
    }
  }
}

// ---------------------------------------------------------------------------
// Row L2-normalize -> packed 16-bit pairs (memory keys; halves the HBM
// stream and the 134 MB result is L2-resident (192 MB) for the sims GEMM)
// ---------------------------------------------------------------------------
__global__ __launch_bounds__(256) void rownorm_e16(const float* __restrict__ in,
                                                   unsigned short* __restrict__ out,
                                                   int cols) {
  __shared__ float red[256];
  const long long base = (long long)blockIdx.x * cols;
  int t = threadIdx.x;
  float ss = 0.f;
  for (int c = t; c < cols; c += 256) { float v = in[base + c]; ss += v * v; }
  red[t] = ss; __syncthreads();
  for (int s = 128; s > 0; s >>= 1) { if (t < s) red[t] += red[t + s]; __syncthreads(); }
  float inv = 1.0f / fmaxf(sqrtf(red[0]), 1e-12f);
  unsigned int* out32 = (unsigned int*)(out + base);
  for (int c = 2 * t; c < cols; c += 512) {
    f32x2 v = *(const f32x2*)(in + base + c);
    out32[c >> 1] = pack2v(v.x * inv, v.y * inv);
  }
}

__global__ __launch_bounds__(256) void rownorm_f32(const float* __restrict__ in,
                                                   float* __restrict__ out, int cols) {
  __shared__ float red[256];
  const long long base = (long long)blockIdx.x * cols;
  int t = threadIdx.x;
  float ss = 0.f;
  for (int c = t; c < cols; c += 256) { float v = in[base + c]; ss += v * v; }
  red[t] = ss; __syncthreads();
  for (int s = 128; s > 0; s >>= 1) { if (t < s) red[t] += red[t + s]; __syncthreads(); }
  float inv = 1.0f / fmaxf(sqrtf(red[0]), 1e-12f);
  for (int c = t; c < cols; c += 256) out[base + c] = in[base + c] * inv;
}

// ---------------------------------------------------------------------------
// out = LayerNorm(x + r) * g + b   (one block per row)
// ---------------------------------------------------------------------------
__global__ __launch_bounds__(256) void add_ln(const float* __restrict__ x,
                                              const float* __restrict__ r,
                                              const float* __restrict__ g,
                                              const float* __restrict__ bta,
                                              float* __restrict__ out, int cols) {
  __shared__ float red[256];
  const long long base = (long long)blockIdx.x * cols;
  int t = threadIdx.x;
  float s1 = 0.f, s2 = 0.f;
  for (int c = t; c < cols; c += 256) {
    float v = x[base + c] + r[base + c];
    s1 += v; s2 += v * v;
  }
  red[t] = s1; __syncthreads();
  for (int s = 128; s > 0; s >>= 1) { if (t < s) red[t] += red[t + s]; __syncthreads(); }
  s1 = red[0]; __syncthreads();
  red[t] = s2; __syncthreads();
  for (int s = 128; s > 0; s >>= 1) { if (t < s) red[t] += red[t + s]; __syncthreads(); }
  s2 = red[0];
  float mean = s1 / cols;
  float var = s2 / cols - mean * mean;
  float inv = rsqrtf(var + 1e-5f);
  for (int c = t; c < cols; c += 256) {
    float v = x[base + c] + r[base + c];
    out[base + c] = (v - mean) * inv * g[c] + bta[c];
  }
}

// ---------------------------------------------------------------------------
// In-place row softmax (one block per row; cols up to 8192)
// ---------------------------------------------------------------------------
__global__ __launch_bounds__(256) void softmax_rows(float* __restrict__ buf, int cols) {
  __shared__ float red[256];
  float* row = buf + (long long)blockIdx.x * cols;
  int t = threadIdx.x;
  float m = -3.0e38f;
  for (int c = t; c < cols; c += 256) m = fmaxf(m, row[c]);
  red[t] = m; __syncthreads();
  for (int s = 128; s > 0; s >>= 1) { if (t < s) red[t] = fmaxf(red[t], red[t + s]); __syncthreads(); }
  m = red[0]; __syncthreads();
  float sum = 0.f;
  for (int c = t; c < cols; c += 256) { float e = expf(row[c] - m); row[c] = e; sum += e; }
  red[t] = sum; __syncthreads();
  for (int s = 128; s > 0; s >>= 1) { if (t < s) red[t] += red[t + s]; __syncthreads(); }
  float inv = 1.0f / red[0];
  for (int c = t; c < cols; c += 256) row[c] *= inv;
}

// ---------------------------------------------------------------------------
// Iterative top-k (k=32) by global argmax with masking. One block per token.
// Ties resolved to the lowest index. Destroys the sims row (workspace-owned).
// ---------------------------------------------------------------------------
__global__ __launch_bounds__(256) void topk_rows(float* __restrict__ sims,
                                                 int* __restrict__ idx,
                                                 int mcols, int kk) {
  __shared__ float bv[256];
  __shared__ int bi[256];
  float* row = sims + (long long)blockIdx.x * mcols;
  int t = threadIdx.x;
  for (int it = 0; it < kk; ++it) {
    float best = -3.0e38f; int besti = 0;
    for (int c = t; c < mcols; c += 256) {
      __builtin_prefetch(row + c + 2048, 0, 0);   // global_prefetch_b8
      float v = row[c];
      if (v > best) { best = v; besti = c; }
    }
    bv[t] = best; bi[t] = besti; __syncthreads();
    for (int s = 128; s > 0; s >>= 1) {
      if (t < s) {
        if (bv[t + s] > bv[t] || (bv[t + s] == bv[t] && bi[t + s] < bi[t])) {
          bv[t] = bv[t + s]; bi[t] = bi[t + s];
        }
      }
      __syncthreads();
    }
    if (t == 0) {
      idx[(long long)blockIdx.x * kk + it] = bi[0];
      row[bi[0]] = -3.0e38f;   // mask selected
    }
    __threadfence_block();
    __syncthreads();
  }
}

// ---------------------------------------------------------------------------
// Gather retrieved value rows: one block per (token,k) row, float4 copies.
// ---------------------------------------------------------------------------
__global__ __launch_bounds__(256) void gather_rows(const float* __restrict__ vals,
                                                   const int* __restrict__ idx,
                                                   float* __restrict__ out, int cols) {
  int r = blockIdx.x;
  int src = idx[r];
  const float4* s = (const float4*)(vals + (long long)src * cols);
  float4* d = (float4*)(out + (long long)r * cols);
  int n4 = cols >> 2;
  for (int c = threadIdx.x; c < n4; c += 256) d[c] = s[c];
}

// ---------------------------------------------------------------------------
// Host orchestration
// ---------------------------------------------------------------------------
extern "C" void kernel_launch(void* const* d_in, const int* in_sizes, int n_in,
                              void* d_out, int out_size, void* d_ws, size_t ws_size,
                              hipStream_t stream) {
  (void)in_sizes; (void)n_in; (void)out_size; (void)ws_size;

  const float* x        = (const float*)d_in[0];
  const float* mem_keys = (const float*)d_in[1];
  const float* mem_vals = (const float*)d_in[2];
  const float* sa_wq = (const float*)d_in[3];
  const float* sa_wk = (const float*)d_in[4];
  const float* sa_wv = (const float*)d_in[5];
  const float* sa_wo = (const float*)d_in[6];
  const float* sa_bq = (const float*)d_in[7];
  const float* sa_bk = (const float*)d_in[8];
  const float* sa_bv = (const float*)d_in[9];
  const float* sa_bo = (const float*)d_in[10];
  const float* ma_wq = (const float*)d_in[11];
  const float* ma_wk = (const float*)d_in[12];
  const float* ma_wv = (const float*)d_in[13];
  const float* ma_wo = (const float*)d_in[14];
  const float* ma_bq = (const float*)d_in[15];
  const float* ma_bk = (const float*)d_in[16];
  const float* ma_bv = (const float*)d_in[17];
  const float* ma_bo = (const float*)d_in[18];
  const float* ln1_g = (const float*)d_in[19];
  const float* ln1_b = (const float*)d_in[20];
  const float* ln2_g = (const float*)d_in[21];
  const float* ln2_b = (const float*)d_in[22];
  const float* ln3_g = (const float*)d_in[23];
  const float* ln3_b = (const float*)d_in[24];
  const float* fc1_w = (const float*)d_in[25];
  const float* fc1_b = (const float*)d_in[26];
  const float* fc2_w = (const float*)d_in[27];
  const float* fc2_b = (const float*)d_in[28];

  // ---- workspace carve (~1.0 GB total) ----
  char* w = (char*)d_ws;
  auto take = [&](size_t bytes) -> void* {
    void* p = (void*)w;
    w += (bytes + 255) & ~(size_t)255;
    return p;
  };
  unsigned short* mkn = (unsigned short*)take((size_t)MM_ * E_ * 2);   // 134 MB
  float* sims  = (float*)take((size_t)NTOK * MM_ * 4);                 // 268 MB
  int*   idxb  = (int*)take((size_t)NTOK * KK_ * 4);
  float* Qb    = (float*)take((size_t)NTOK * E_ * 4);
  float* Kb    = (float*)take((size_t)NTOK * E_ * 4);
  float* Vb    = (float*)take((size_t)NTOK * E_ * 4);
  float* x1    = (float*)take((size_t)NTOK * E_ * 4);
  float* x2    = (float*)take((size_t)NTOK * E_ * 4);
  float* qn    = (float*)take((size_t)NTOK * E_ * 4);
  float* ao    = (float*)take((size_t)NTOK * E_ * 4);
  float* proj  = (float*)take((size_t)NTOK * E_ * 4);
  float* scS   = (float*)take((size_t)B_ * H_ * S_ * S_ * 4);          // 16.8 MB
  float* rflat = (float*)take((size_t)NTOK * KK_ * E_ * 4);            // 134 MB
  float* Kp    = (float*)take((size_t)NTOK * KK_ * E_ * 4);            // 134 MB
  float* Vp    = (float*)take((size_t)NTOK * KK_ * E_ * 4);            // 134 MB
  float* scC   = (float*)take((size_t)H_ * S_ * SK_ * 4);              // 134 MB (per-b reuse)
  float* hb    = (float*)take((size_t)NTOK * FF_ * 4);                 // 16.8 MB

  const float inv_sqrt_hd = 0.125f;  // 1/sqrt(64)
  const long long SE = (long long)S_ * E_;
  const long long SS = (long long)S_ * S_;

  // 0) normalized memory keys -> bf16 (L2-resident stream for sims GEMM)
  rownorm_e16<<<MM_, 256, 0, stream>>>(mem_keys, mkn, E_);

  // 1) self-attention QKV projections: (1024xE)x(ExE)+b
  dim3 gP(E_ / TN, NTOK / TM, 1);
  gemm_wmma<false, float, 0><<<gP, 256, 0, stream>>>(x, E_, 0, 0, sa_wq, E_, 0, 0, Qb, E_, 0, 0, NTOK, E_, E_, 1, sa_bq, 1.0f);
  gemm_wmma<false, float, 0><<<gP, 256, 0, stream>>>(x, E_, 0, 0, sa_wk, E_, 0, 0, Kb, E_, 0, 0, NTOK, E_, E_, 1, sa_bk, 1.0f);
  gemm_wmma<false, float, 0><<<gP, 256, 0, stream>>>(x, E_, 0, 0, sa_wv, E_, 0, 0, Vb, E_, 0, 0, NTOK, E_, E_, 1, sa_bv, 1.0f);

  // 2) scores[b,h] = Q_bh * K_bh^T / 8   (z = b*H+h)
  gemm_wmma<true, float, 0><<<dim3(S_ / TN, S_ / TM, B_ * H_), 256, 0, stream>>>(
      Qb, E_, SE, HD_, Kb, E_, SE, HD_, scS, S_, (long long)H_ * SS, SS,
      S_, S_, HD_, H_, nullptr, inv_sqrt_hd);

  // 3) softmax over keys
  softmax_rows<<<B_ * H_ * S_, 256, 0, stream>>>(scS, S_);

  // 4) out[b,:,h,:] = attn * V_bh  (Nd = 64: N-guards handle the half tile)
  gemm_wmma<false, float, 0><<<dim3(1, S_ / TM, B_ * H_), 256, 0, stream>>>(
      scS, S_, (long long)H_ * SS, SS, Vb, E_, SE, HD_, ao, E_, SE, HD_,
      S_, HD_, S_, H_, nullptr, 1.0f);

  // 5) output projection + 6) LN1
  gemm_wmma<false, float, 0><<<gP, 256, 0, stream>>>(ao, E_, 0, 0, sa_wo, E_, 0, 0, proj, E_, 0, 0, NTOK, E_, E_, 1, sa_bo, 1.0f);
  add_ln<<<NTOK, 256, 0, stream>>>(x, proj, ln1_g, ln1_b, x1, E_);

  // 7) query normalize, 8) cosine sims = qn * mkn^T  (the 137 GFLOP GEMM)
  rownorm_f32<<<NTOK, 256, 0, stream>>>(x1, qn, E_);
  gemm_wmma<true, unsigned short, 0><<<dim3(MM_ / TN, NTOK / TM, 1), 256, 0, stream>>>(
      qn, E_, 0, 0, mkn, E_, 0, 0, sims, MM_, 0, 0, NTOK, MM_, E_, 1, nullptr, 1.0f);

  // 9) top-32 per token, 10) gather values
  topk_rows<<<NTOK, 256, 0, stream>>>(sims, idxb, MM_, KK_);
  gather_rows<<<NTOK * KK_, 256, 0, stream>>>(mem_vals, idxb, rflat, E_);

  // 11) cross-attn projections: Qc from x1; Kp/Vp from 32768 gathered rows
  gemm_wmma<false, float, 0><<<gP, 256, 0, stream>>>(x1, E_, 0, 0, ma_wq, E_, 0, 0, Qb, E_, 0, 0, NTOK, E_, E_, 1, ma_bq, 1.0f);
  dim3 gR(E_ / TN, (NTOK * KK_) / TM, 1);
  gemm_wmma<false, float, 0><<<gR, 256, 0, stream>>>(rflat, E_, 0, 0, ma_wk, E_, 0, 0, Kp, E_, 0, 0, NTOK * KK_, E_, E_, 1, ma_bk, 1.0f);
  gemm_wmma<false, float, 0><<<gR, 256, 0, stream>>>(rflat, E_, 0, 0, ma_wv, E_, 0, 0, Vp, E_, 0, 0, NTOK * KK_, E_, E_, 1, ma_bv, 1.0f);

  // 12) cross-attention, chunked per batch b (scores buffer reused: 134 MB)
  for (int b = 0; b < B_; ++b) {
    const float* Qcb = Qb + (long long)b * SE;
    const float* Kpb = Kp + (long long)b * SK_ * E_;
    const float* Vpb = Vp + (long long)b * SK_ * E_;
    float* aob = ao + (long long)b * SE;
    gemm_wmma<true, float, 0><<<dim3(SK_ / TN, S_ / TM, H_), 256, 0, stream>>>(
        Qcb, E_, 0, HD_, Kpb, E_, 0, HD_, scC, SK_, 0, (long long)S_ * SK_,
        S_, SK_, HD_, H_, nullptr, inv_sqrt_hd);
    softmax_rows<<<H_ * S_, 256, 0, stream>>>(scC, SK_);
    gemm_wmma<false, float, 0><<<dim3(1, S_ / TM, H_), 256, 0, stream>>>(
        scC, SK_, 0, (long long)S_ * SK_, Vpb, E_, 0, HD_, aob, E_, 0, HD_,
        S_, HD_, SK_, H_, nullptr, 1.0f);
  }

  // 13) cross output projection + 14) LN2
  gemm_wmma<false, float, 0><<<gP, 256, 0, stream>>>(ao, E_, 0, 0, ma_wo, E_, 0, 0, proj, E_, 0, 0, NTOK, E_, E_, 1, ma_bo, 1.0f);
  add_ln<<<NTOK, 256, 0, stream>>>(x1, proj, ln2_g, ln2_b, x2, E_);

  // 15) FFN: fc1 + exact GELU fused epilogue, then fc2
  gemm_wmma<false, float, 1><<<dim3(FF_ / TN, NTOK / TM, 1), 256, 0, stream>>>(
      x2, E_, 0, 0, fc1_w, FF_, 0, 0, hb, FF_, 0, 0, NTOK, FF_, E_, 1, fc1_b, 1.0f);
  gemm_wmma<false, float, 0><<<gP, 256, 0, stream>>>(
      hb, FF_, 0, 0, fc2_w, E_, 0, 0, proj, E_, 0, 0, NTOK, E_, FF_, 1, fc2_b, 1.0f);

  // 16) LN3 -> final output
  add_ln<<<NTOK, 256, 0, stream>>>(x2, proj, ln3_g, ln3_b, (float*)d_out, E_);
}
